// RAWOpticalFlow_45397804318957
// MI455X (gfx1250) — compile-verified, matrix operand order checked by reading, not run on hardware
//
#include <hip/hip_runtime.h>
#include <math.h>

typedef _Float16 f16;
typedef __attribute__((ext_vector_type(8)))  f16   v8h;
typedef __attribute__((ext_vector_type(16))) f16   v16h;
typedef __attribute__((ext_vector_type(8)))  float v8f;

#define ACT_NONE 0
#define ACT_RELU 1
#define ACT_SIG  2
#define ACT_TANH 3

#ifndef __has_builtin
#define __has_builtin(x) 0
#endif

// gfx1250 async global->LDS copy path (ASYNCcnt-tracked), with fallback.
#if defined(__gfx1250__) && __has_builtin(__builtin_amdgcn_global_load_async_to_lds_b128)
#define USE_ASYNC_LDS 1
// Builtin signature (from hipcc diagnostic): takes 4xi32-vector pointers,
// global side in address_space(1) (printed "__device__"), LDS side in
// address_space(3).
typedef int v4i_ __attribute__((vector_size(16)));
typedef __attribute__((address_space(1))) v4i_ as1_v4i;
typedef __attribute__((address_space(3))) v4i_ as3_v4i;
#else
#define USE_ASYNC_LDS 0
#endif

static inline int align_up_i(int x, int a) { return (x + a - 1) / a * a; }

// ---------------------------------------------------------------------------
// WMMA implicit-GEMM:  out = act(A[M,Kp] * W[N,Kp]^T + bias)
// A, W are f16 row-major with Kp (K padded to 32) zero-padded columns.
//
// Workgroup = 4 waves (128 threads). blockIdx.x selects a 16-col N tile,
// blockIdx.y a 128-row M slab; wave w computes a 32x16 sub-slab (2 WMMA
// accumulators). The 16x32 f16 B tile is staged once per k-step into LDS
// (async global->LDS on gfx1250; ds_store fallback) and read back with two
// ds b128 loads per lane, so all 8 WMMA tiles in the block share one B fetch.
//
// Fragment layout per CDNA5 ISA 7.12.2 (16-bit A 16x32):
//   lane = h*16 + r (h = lane>>4, r = lane&15); row r holds
//   K[k0+8h .. k0+8h+7] and K[k0+16+8h .. k0+16+8h+7] -> two b128 loads.
// ---------------------------------------------------------------------------
__global__ __launch_bounds__(128) void wmma_gemm_k(
    const f16* __restrict__ A, const f16* __restrict__ W,
    const float* __restrict__ bias, float* __restrict__ out,
    int M, int N, int Kp, int act, int out_mn, int ldout)
{
  __shared__ __align__(16) f16 Bs[16 * 32];   // one 16(N) x 32(K) tile
  const int tid   = threadIdx.x;
  const int wv    = tid >> 5;
  const int lane  = tid & 31;
  const int half_ = lane >> 4;
  const int r16   = lane & 15;
  const int mbase = blockIdx.y * 128 + wv * 32;
  const f16* Arow0 = A + (size_t)(mbase + r16) * Kp;
  const f16* Arow1 = A + (size_t)(mbase + 16 + r16) * Kp;
  const f16* Wtile = W + (size_t)blockIdx.x * 16 * Kp;
  v8f acc0 = {};
  v8f acc1 = {};
  for (int k0 = 0; k0 < Kp; k0 += 32) {
    // --- stage B tile into LDS (64 lanes x 16B = 1KB) ---
    if (tid < 64) {
      int n = tid >> 2, q = tid & 3;
      const f16* g = Wtile + (size_t)n * Kp + k0 + q * 8;
      f16* l = Bs + n * 32 + q * 8;
#if USE_ASYNC_LDS
      __builtin_amdgcn_global_load_async_to_lds_b128((as1_v4i*)g, (as3_v4i*)l, 0, 0);
#else
      *(v8h*)l = *(const v8h*)g;
#endif
    }
#if USE_ASYNC_LDS
#if __has_builtin(__builtin_amdgcn_s_wait_asynccnt)
    __builtin_amdgcn_s_wait_asynccnt(0);
#else
    asm volatile("s_wait_asynccnt 0" ::: "memory");
#endif
#endif
    __syncthreads();
    // --- B fragment from LDS (shared by all 4 waves) ---
    v8h b0 = *(const v8h*)(Bs + r16 * 32 + 8 * half_);
    v8h b1 = *(const v8h*)(Bs + r16 * 32 + 16 + 8 * half_);
    v16h bv = __builtin_shufflevector(b0, b1, 0,1,2,3,4,5,6,7,8,9,10,11,12,13,14,15);
    // --- A fragments from global (CDNA5 prefetch 8 k-tiles ahead) ---
    __builtin_prefetch(Arow0 + k0 + 256, 0, 0);
    __builtin_prefetch(Arow1 + k0 + 256, 0, 0);
    v8h a00 = *(const v8h*)(Arow0 + k0 + 8 * half_);
    v8h a01 = *(const v8h*)(Arow0 + k0 + 16 + 8 * half_);
    v8h a10 = *(const v8h*)(Arow1 + k0 + 8 * half_);
    v8h a11 = *(const v8h*)(Arow1 + k0 + 16 + 8 * half_);
    v16h av0 = __builtin_shufflevector(a00, a01, 0,1,2,3,4,5,6,7,8,9,10,11,12,13,14,15);
    v16h av1 = __builtin_shufflevector(a10, a11, 0,1,2,3,4,5,6,7,8,9,10,11,12,13,14,15);
    acc0 = __builtin_amdgcn_wmma_f32_16x16x32_f16(false, av0, false, bv,
                                                  (short)0, acc0, false, false);
    acc1 = __builtin_amdgcn_wmma_f32_16x16x32_f16(false, av1, false, bv,
                                                  (short)0, acc1, false, false);
    __syncthreads();   // protect Bs before next k-step's fill
  }
  // C/D layout: lane l -> n = l&15; VGPR v -> m(tile) = 8*(l>>4) + v
  const int n = blockIdx.x * 16 + r16;
  if (n >= N) return;  // after all WMMAs/barriers: EXEC all-ones during matrix ops
  const float bv_ = bias ? bias[n] : 0.0f;
#pragma unroll
  for (int v = 0; v < 8; ++v) {
    int m0 = mbase + 8 * half_ + v;
    int m1 = mbase + 16 + 8 * half_ + v;
    float x0 = acc0[v] + bv_;
    float x1 = acc1[v] + bv_;
    if (act == ACT_RELU)      { x0 = fmaxf(x0, 0.0f); x1 = fmaxf(x1, 0.0f); }
    else if (act == ACT_SIG)  { x0 = 1.0f / (1.0f + expf(-x0)); x1 = 1.0f / (1.0f + expf(-x1)); }
    else if (act == ACT_TANH) { x0 = tanhf(x0); x1 = tanhf(x1); }
    if (out_mn) {
      out[(size_t)m0 * ldout + n] = x0;
      out[(size_t)m1 * ldout + n] = x1;
    } else {
      out[(size_t)n * ldout + m0] = x0;
      out[(size_t)n * ldout + m1] = x1;
    }
  }
}

// ---------------------------------------------------------------------------
// im2col with up to 4 channel-concatenated source maps ([C][Hi][Wi] f32 each)
// -> A[M = Ho*Wo][Kp] f16, zero-padded for k >= K and out-of-image taps.
// ---------------------------------------------------------------------------
__global__ void im2col_k(
    const float* __restrict__ p0, int c0, const float* __restrict__ p1, int c1,
    const float* __restrict__ p2, int c2, const float* __restrict__ p3, int c3,
    int Hi, int Wi, int Ho, int Wo, int ksz, int stride, int pad,
    int K, int Kp, f16* __restrict__ A)
{
  long long idx = (long long)blockIdx.x * blockDim.x + threadIdx.x;
  long long total = (long long)Ho * Wo * Kp;
  if (idx >= total) return;
  int kk = (int)(idx % Kp);
  int p  = (int)(idx / Kp);
  float val = 0.0f;
  if (kk < K) {
    int kk2 = ksz * ksz;
    int c  = kk / kk2;
    int rr = kk - c * kk2;
    int dy = rr / ksz, dx = rr - dy * ksz;
    int y = p / Wo, x = p - y * Wo;
    int iy = y * stride - pad + dy;
    int ix = x * stride - pad + dx;
    if (iy >= 0 && iy < Hi && ix >= 0 && ix < Wi) {
      const float* src; int cl = c;
      if (cl < c0) { src = p0; }
      else { cl -= c0;
        if (cl < c1) { src = p1; }
        else { cl -= c1;
          if (cl < c2) { src = p2; }
          else { cl -= c2; src = p3; } } }
      val = src[((size_t)cl * Hi + iy) * Wi + ix];
    }
  }
  A[idx] = (f16)val;
}

// Pack conv weights [N][K] f32 -> [Np][Kp] f16 with zero padding.
__global__ void packw_k(const float* __restrict__ w, f16* __restrict__ Wp,
                        int N, int K, int Np, int Kp)
{
  int idx = blockIdx.x * blockDim.x + threadIdx.x;
  if (idx >= Np * Kp) return;
  int k = idx % Kp, o = idx / Kp;
  float v = (o < N && k < K) ? w[(size_t)o * K + k] : 0.0f;
  Wp[idx] = (f16)v;
}

// Instance norm over one channel per block (LDS tree reduction), optional relu.
__global__ __launch_bounds__(256) void inorm_k(float* __restrict__ x, int HW, int relu)
{
  __shared__ float ssum[256];
  __shared__ float ssq[256];
  float* ch = x + (size_t)blockIdx.x * HW;
  int tid = threadIdx.x;
  float s = 0.f, q = 0.f;
  for (int i = tid; i < HW; i += 256) { float v = ch[i]; s += v; q += v * v; }
  ssum[tid] = s; ssq[tid] = q;
  __syncthreads();
  for (int st = 128; st > 0; st >>= 1) {
    if (tid < st) { ssum[tid] += ssum[tid + st]; ssq[tid] += ssq[tid + st]; }
    __syncthreads();
  }
  float mean = ssum[0] / (float)HW;
  float var  = ssq[0] / (float)HW - mean * mean;
  float inv  = rsqrtf(var + 1e-5f);
  for (int i = tid; i < HW; i += 256) {
    float v = (ch[i] - mean) * inv;
    if (relu) v = fmaxf(v, 0.f);
    ch[i] = v;
  }
}

// ---- pointwise kernels ----
__global__ void ew_addrelu_k(float* a, const float* __restrict__ b, int n)
{ int i = blockIdx.x * 256 + threadIdx.x; if (i < n) a[i] = fmaxf(a[i] + b[i], 0.f); }
__global__ void ew_addto_k(float* a, const float* __restrict__ b, int n)
{ int i = blockIdx.x * 256 + threadIdx.x; if (i < n) a[i] += b[i]; }
__global__ void ew_tanh_k(const float* __restrict__ in, float* out, int n)
{ int i = blockIdx.x * 256 + threadIdx.x; if (i < n) out[i] = tanhf(in[i]); }
__global__ void ew_mul_k(const float* __restrict__ a, const float* __restrict__ b, float* o, int n)
{ int i = blockIdx.x * 256 + threadIdx.x; if (i < n) o[i] = a[i] * b[i]; }
__global__ void ew_gru_k(float* h, const float* __restrict__ z, const float* __restrict__ q, int n)
{ int i = blockIdx.x * 256 + threadIdx.x; if (i < n) h[i] = (1.f - z[i]) * h[i] + z[i] * q[i]; }
__global__ void fill_k(float* p, float v, int n)
{ int i = blockIdx.x * 256 + threadIdx.x; if (i < n) p[i] = v; }

// Per-pixel L2 normalize over channels + transpose [C][HW] f32 -> [HW][C] f16
__global__ void normt_k(const float* __restrict__ f, f16* __restrict__ ft, int C, int HW)
{
  int p = blockIdx.x * 256 + threadIdx.x;
  if (p >= HW) return;
  float s = 0.f;
  for (int c = 0; c < C; ++c) { float v = f[(size_t)c * HW + p]; s += v * v; }
  float inv = 1.0f / (sqrtf(s) + 1e-8f);
  for (int c = 0; c < C; ++c) ft[(size_t)p * C + c] = (f16)(f[(size_t)c * HW + p] * inv);
}

// 2x2 average pool over last two dims of [N][h][w]
__global__ void pool_k(const float* __restrict__ in, float* __restrict__ out,
                       int N, int h, int w)
{
  int ho = h >> 1, wo = w >> 1;
  long long idx = (long long)blockIdx.x * 256 + threadIdx.x;
  long long total = (long long)N * ho * wo;
  if (idx >= total) return;
  int x = (int)(idx % wo);
  long long t = idx / wo;
  int y = (int)(t % ho);
  int n = (int)(t / ho);
  const float* b = in + (size_t)n * h * w;
  out[idx] = 0.25f * (b[(2*y)*w + 2*x]   + b[(2*y)*w + 2*x+1] +
                      b[(2*y+1)*w + 2*x] + b[(2*y+1)*w + 2*x+1]);
}

// RAFT multi-level correlation window lookup; out[324][1024]
__global__ void corr_gather_k(const float* __restrict__ py0, const float* __restrict__ py1,
                              const float* __restrict__ py2, const float* __restrict__ py3,
                              const float* __restrict__ flow, float* __restrict__ out)
{
  int idx = blockIdx.x * 256 + threadIdx.x;
  if (idx >= 324 * 1024) return;
  int p = idx & 1023, c = idx >> 10;
  int level = c / 81, j = c - level * 81;
  float dx = (float)(j % 9) - 4.0f;
  float dy = (float)(j / 9) - 4.0f;
  int x = p & 31, y = p >> 5;
  float cx = (float)x + flow[p];
  float cy = (float)y + flow[1024 + p];
  float s = (float)(1 << level);
  float xs = cx / s + dx, ys = cy / s + dy;
  const float* img; int w;
  if (level == 0)      { img = py0 + (size_t)p * 1024; w = 32; }
  else if (level == 1) { img = py1 + (size_t)p * 256;  w = 16; }
  else if (level == 2) { img = py2 + (size_t)p * 64;   w = 8;  }
  else                 { img = py3 + (size_t)p * 16;   w = 4;  }
  float x0f = floorf(xs), y0f = floorf(ys);
  float wx = xs - x0f, wy = ys - y0f;
  int x0 = (int)x0f, y0 = (int)y0f;
  bool xa = (x0 >= 0) && (x0 <= w - 1);
  bool xb = (x0 + 1 >= 0) && (x0 + 1 <= w - 1);
  bool ya = (y0 >= 0) && (y0 <= w - 1);
  bool yb = (y0 + 1 >= 0) && (y0 + 1 <= w - 1);
  float v00 = (ya && xa) ? img[y0 * w + x0]           : 0.f;
  float v01 = (ya && xb) ? img[y0 * w + x0 + 1]       : 0.f;
  float v10 = (yb && xa) ? img[(y0 + 1) * w + x0]     : 0.f;
  float v11 = (yb && xb) ? img[(y0 + 1) * w + x0 + 1] : 0.f;
  out[idx] = (1.f - wy) * (1.f - wx) * v00 + (1.f - wy) * wx * v01
           + wy * (1.f - wx) * v10 + wy * wx * v11;
}

// Bilinear resize (half-pixel centers, edge clamp), out = mult * interp(in)
__global__ void resize_k(const float* __restrict__ in, float* __restrict__ out,
                         int C, int Hi, int Wi, int Ho, int Wo, float mult)
{
  long long idx = (long long)blockIdx.x * 256 + threadIdx.x;
  long long total = (long long)C * Ho * Wo;
  if (idx >= total) return;
  int x = (int)(idx % Wo);
  long long t = idx / Wo;
  int y = (int)(t % Ho);
  int c = (int)(t / Ho);
  float sy = ((float)y + 0.5f) * (float)Hi / (float)Ho - 0.5f;
  float sx = ((float)x + 0.5f) * (float)Wi / (float)Wo - 0.5f;
  sy = fminf(fmaxf(sy, 0.f), (float)(Hi - 1));
  sx = fminf(fmaxf(sx, 0.f), (float)(Wi - 1));
  int y0 = (int)sy, x0 = (int)sx;
  int y1 = (y0 + 1 < Hi) ? y0 + 1 : Hi - 1;
  int x1 = (x0 + 1 < Wi) ? x0 + 1 : Wi - 1;
  float wy = sy - (float)y0, wx = sx - (float)x0;
  const float* b = in + (size_t)c * Hi * Wi;
  float v = (1.f - wy) * ((1.f - wx) * b[y0 * Wi + x0] + wx * b[y0 * Wi + x1])
          + wy * ((1.f - wx) * b[y1 * Wi + x0] + wx * b[y1 * Wi + x1]);
  out[idx] = mult * v;
}

// ---------------------------------------------------------------------------
// Host-side sequencing
// ---------------------------------------------------------------------------
struct Ctx { hipStream_t st; f16* Abuf; f16* Wbuf; };
struct RB  { const float *b1,*b2,*b3,*bs,*w1,*w2,*w3,*ws; };

static void launch_conv(const Ctx& C,
    const float* s0, int c0, const float* s1, int c1,
    const float* s2, int c2, const float* s3, int c3,
    int Hi, int Wi, int O, int ksz, int stride, int pad,
    const float* w, const float* b, float* out, int act)
{
  int Ho = (Hi + 2 * pad - ksz) / stride + 1;
  int Wo = (Wi + 2 * pad - ksz) / stride + 1;
  int Ct = c0 + c1 + c2 + c3;
  int K  = Ct * ksz * ksz;
  int Kp = align_up_i(K, 32);
  int Np = align_up_i(O, 16);
  int M  = Ho * Wo;
  long long tot = (long long)M * Kp;
  im2col_k<<<(unsigned)((tot + 255) / 256), 256, 0, C.st>>>(
      s0, c0, s1, c1, s2, c2, s3, c3, Hi, Wi, Ho, Wo, ksz, stride, pad, K, Kp, C.Abuf);
  int wt = Np * Kp;
  packw_k<<<(wt + 255) / 256, 256, 0, C.st>>>(w, C.Wbuf, O, K, Np, Kp);
  dim3 g((unsigned)(Np / 16), (unsigned)(M / 128));
  wmma_gemm_k<<<g, 128, 0, C.st>>>(C.Abuf, C.Wbuf, b, out, M, O, Kp, act, 0, M);
}

static void conv1s(const Ctx& C, const float* x, int Ci, int Hi, int Wi,
                   int O, int k, int s, int p, const float* w, const float* b,
                   float* out, int act)
{ launch_conv(C, x, Ci, nullptr, 0, nullptr, 0, nullptr, 0, Hi, Wi, O, k, s, p, w, b, out, act); }

static void resblock(const Ctx& C, const RB& P, const float* x, int Ci, int Hi, int Wi,
                     int Co, int stride, float* out, float* t1, float* t2, float* res)
{
  int Cm = Co / 2;
  int Ho = (Hi + 2 - 3) / stride + 1;
  int Wo = (Wi + 2 - 3) / stride + 1;
  conv1s(C, x, Ci, Hi, Wi, Cm, 1, 1, 0, P.w1, P.b1, t1, ACT_NONE);
  inorm_k<<<Cm, 256, 0, C.st>>>(t1, Hi * Wi, 1);
  conv1s(C, t1, Cm, Hi, Wi, Cm, 3, stride, 1, P.w2, P.b2, t2, ACT_NONE);
  inorm_k<<<Cm, 256, 0, C.st>>>(t2, Ho * Wo, 1);
  conv1s(C, t2, Cm, Ho, Wo, Co, 1, 1, 0, P.w3, P.b3, out, ACT_NONE);
  inorm_k<<<Co, 256, 0, C.st>>>(out, Ho * Wo, 0);
  int n = Co * Ho * Wo;
  if (P.ws) {
    conv1s(C, x, Ci, Hi, Wi, Co, 1, stride, 0, P.ws, P.bs, res, ACT_NONE);
    inorm_k<<<Co, 256, 0, C.st>>>(res, Ho * Wo, 0);
    ew_addrelu_k<<<(n + 255) / 256, 256, 0, C.st>>>(out, res, n);
  } else {
    ew_addrelu_k<<<(n + 255) / 256, 256, 0, C.st>>>(out, x, n);
  }
}

extern "C" void kernel_launch(void* const* d_in, const int* in_sizes, int n_in,
                              void* d_out, int out_size, void* d_ws, size_t ws_size,
                              hipStream_t stream)
{
  (void)in_sizes; (void)n_in; (void)out_size; (void)ws_size;
  const float* ref_img = (const float*)d_in[0];
  const float* tgt_img = (const float*)d_in[1];

  // ---- parameter leaves in jax pytree order (sorted dict keys per level) ----
  int pi = 2;
  auto nx = [&]() { return (const float*)d_in[pi++]; };
  // conf
  const float *conf_c1b = nx(), *conf_c1w = nx();
  const float *conf_c2b = nx(), *conf_c2w = nx();
  const float *conf_c3b = nx(), *conf_c3w = nx();
  // ctx
  const float *cx_c1b = nx(), *cx_c1w = nx();
  const float *cx_c2b = nx(), *cx_c2w = nx();
  const float *cx_c3b = nx(), *cx_c3w = nx();
  const float *cx_c4b = nx(), *cx_c4w = nx();
  const float *cx_cob = nx(), *cx_cow = nx();
  // feat
  const float *ft_c1b = nx(), *ft_c1w = nx();
  auto rb_plain = [&]() { RB r{}; r.b1=nx(); r.b2=nx(); r.b3=nx(); r.w1=nx(); r.w2=nx(); r.w3=nx(); return r; };
  auto rb_short = [&]() { RB r{}; r.b1=nx(); r.b2=nx(); r.b3=nx(); r.bs=nx(); r.w1=nx(); r.w2=nx(); r.w3=nx(); r.ws=nx(); return r; };
  RB l1a = rb_plain(), l1b = rb_plain();
  RB l2a = rb_short(), l2b = rb_plain();
  RB l3a = rb_short(), l3b = rb_plain();
  RB l4a = rb_short(), l4b = rb_plain();
  const float *ft_pb = nx(), *ft_pw = nx();
  // upd
  const float *e1b = nx(), *e1w = nx(), *e2b = nx(), *e2w = nx();
  const float *f1b = nx(), *f1w = nx(), *f2b = nx(), *f2w = nx();
  const float *qb  = nx(), *qw  = nx(), *rbb = nx(), *rww = nx();
  const float *zb  = nx(), *zw  = nx();

  // ---- workspace bump allocator ----
  char* base = (char*)d_ws;
  size_t off = 0;
  auto alloc = [&](size_t bytes) -> void* {
    void* r = base + off; off += (bytes + 255) & ~(size_t)255; return r;
  };
  f16*   Abuf = (f16*)alloc((size_t)65536 * 288 * sizeof(f16));  // max M*Kp
  f16*   Wbuf = (f16*)alloc((size_t)128 * 6976 * sizeof(f16));   // max Np*Kp
  float* X    = (float*)alloc((size_t)64 * 65536 * 4);
  float* Y    = (float*)alloc((size_t)64 * 65536 * 4);
  float* T1   = (float*)alloc((size_t)48 * 65536 * 4);
  float* T2   = (float*)alloc((size_t)32 * 65536 * 4);
  float* Rb   = (float*)alloc((size_t)32 * 65536 * 4);
  float* f1m  = (float*)alloc((size_t)128 * 1024 * 4);
  float* f2m  = (float*)alloc((size_t)128 * 1024 * 4);
  float* ctxm = (float*)alloc((size_t)256 * 1024 * 4);
  f16*   f1t  = (f16*)alloc((size_t)1024 * 128 * sizeof(f16));
  f16*   f2t  = (f16*)alloc((size_t)1024 * 128 * sizeof(f16));
  float* py0  = (float*)alloc((size_t)1024 * 1024 * 4);
  float* py1  = (float*)alloc((size_t)1024 * 256 * 4);
  float* py2  = (float*)alloc((size_t)1024 * 64 * 4);
  float* py3  = (float*)alloc((size_t)1024 * 16 * 4);
  float* cfeat= (float*)alloc((size_t)324 * 1024 * 4);
  float* hid  = (float*)alloc((size_t)128 * 1024 * 4);
  float* zbuf = (float*)alloc((size_t)128 * 1024 * 4);
  float* rbuf = (float*)alloc((size_t)128 * 1024 * 4);
  float* qbuf = (float*)alloc((size_t)128 * 1024 * 4);
  float* rh   = (float*)alloc((size_t)128 * 1024 * 4);
  float* mt   = (float*)alloc((size_t)128 * 1024 * 4);
  float* mb   = (float*)alloc((size_t)64 * 1024 * 4);
  float* flow = (float*)alloc((size_t)2 * 1024 * 4);
  float* df   = (float*)alloc((size_t)2 * 1024 * 4);
  float* cb1  = (float*)alloc((size_t)64 * 1024 * 4);
  float* cb2  = (float*)alloc((size_t)32 * 1024 * 4);
  float* cb3  = (float*)alloc((size_t)1024 * 4);

  Ctx C{stream, Abuf, Wbuf};

  // ---- feature encoder (shared weights) on both images ----
  const float* imgs[2] = {ref_img, tgt_img};
  float* fouts[2] = {f1m, f2m};
  for (int im = 0; im < 2; ++im) {
    conv1s(C, imgs[im], 4, 256, 256, 64, 7, 1, 3, ft_c1w, ft_c1b, X, ACT_NONE);
    inorm_k<<<64, 256, 0, stream>>>(X, 65536, 1);
    resblock(C, l1a, X, 64, 256, 256, 64, 1, Y, T1, T2, Rb);
    resblock(C, l1b, Y, 64, 256, 256, 64, 1, X, T1, T2, Rb);
    resblock(C, l2a, X, 64, 256, 256, 96, 2, Y, T1, T2, Rb);
    resblock(C, l2b, Y, 96, 128, 128, 96, 1, X, T1, T2, Rb);
    resblock(C, l3a, X, 96, 128, 128, 128, 2, Y, T1, T2, Rb);
    resblock(C, l3b, Y, 128, 64, 64, 128, 1, X, T1, T2, Rb);
    resblock(C, l4a, X, 128, 64, 64, 128, 2, Y, T1, T2, Rb);
    resblock(C, l4b, Y, 128, 32, 32, 128, 1, X, T1, T2, Rb);
    conv1s(C, X, 128, 32, 32, 128, 1, 1, 0, ft_pw, ft_pb, fouts[im], ACT_NONE);
  }

  // ---- context encoder (relu fused into GEMM epilogue) ----
  conv1s(C, ref_img, 4, 256, 256, 64, 7, 1, 3, cx_c1w, cx_c1b, X, ACT_RELU);
  conv1s(C, X, 64, 256, 256, 64, 3, 2, 1, cx_c2w, cx_c2b, Y, ACT_RELU);
  conv1s(C, Y, 64, 128, 128, 96, 3, 2, 1, cx_c3w, cx_c3b, X, ACT_RELU);
  conv1s(C, X, 96, 64, 64, 128, 3, 2, 1, cx_c4w, cx_c4b, Y, ACT_RELU);
  conv1s(C, Y, 128, 32, 32, 256, 1, 1, 0, cx_cow, cx_cob, ctxm, ACT_NONE);

  // hidden = tanh(ctx[:128]); flow = 0
  ew_tanh_k<<<(128 * 1024 + 255) / 256, 256, 0, stream>>>(ctxm, hid, 128 * 1024);
  fill_k<<<(2048 + 255) / 256, 256, 0, stream>>>(flow, 0.f, 2048);

  // ---- correlation pyramid: normalize, transpose, WMMA GEMM, pool ----
  normt_k<<<4, 256, 0, stream>>>(f1m, f1t, 128, 1024);
  normt_k<<<4, 256, 0, stream>>>(f2m, f2t, 128, 1024);
  {
    dim3 g(64, 8);
    wmma_gemm_k<<<g, 128, 0, stream>>>(f1t, f2t, nullptr, py0,
                                       1024, 1024, 128, ACT_NONE, 1, 1024);
  }
  pool_k<<<(1024 * 16 * 16 + 255) / 256, 256, 0, stream>>>(py0, py1, 1024, 32, 32);
  pool_k<<<(1024 * 8 * 8 + 255) / 256, 256, 0, stream>>>(py1, py2, 1024, 16, 16);
  pool_k<<<(1024 * 4 * 4 + 255) / 256, 256, 0, stream>>>(py2, py3, 1024, 8, 8);

  // ---- 12 GRU update iterations ----
  const int HW = 1024, NP = 128 * 1024;
  for (int it = 0; it < 12; ++it) {
    corr_gather_k<<<(324 * 1024 + 255) / 256, 256, 0, stream>>>(py0, py1, py2, py3, flow, cfeat);
    // motion encoder: e1 over [corr|ctx|flow], e2 over e1
    launch_conv(C, cfeat, 324, ctxm, 256, flow, 2, nullptr, 0,
                32, 32, 128, 3, 1, 1, e1w, e1b, mt, ACT_RELU);
    conv1s(C, mt, 128, 32, 32, 64, 3, 1, 1, e2w, e2b, mb, ACT_RELU);
    // gates over hx = [hidden|corr|ctx|m]
    launch_conv(C, hid, 128, cfeat, 324, ctxm, 256, mb, 64,
                32, 32, 128, 3, 1, 1, zw, zb, zbuf, ACT_SIG);
    launch_conv(C, hid, 128, cfeat, 324, ctxm, 256, mb, 64,
                32, 32, 128, 3, 1, 1, rww, rbb, rbuf, ACT_SIG);
    ew_mul_k<<<(NP + 255) / 256, 256, 0, stream>>>(rbuf, hid, rh, NP);
    launch_conv(C, rh, 128, cfeat, 324, ctxm, 256, mb, 64,
                32, 32, 128, 3, 1, 1, qw, qb, qbuf, ACT_TANH);
    ew_gru_k<<<(NP + 255) / 256, 256, 0, stream>>>(hid, zbuf, qbuf, NP);
    // flow head
    conv1s(C, hid, 128, 32, 32, 128, 3, 1, 1, f1w, f1b, mt, ACT_RELU);
    conv1s(C, mt, 128, 32, 32, 2, 3, 1, 1, f2w, f2b, df, ACT_NONE);
    ew_addto_k<<<(2 * HW + 255) / 256, 256, 0, stream>>>(flow, df, 2 * HW);
  }

  // ---- outputs: 8 * upsample(flow), confidence head + upsample ----
  float* out_flow = (float*)d_out;
  float* out_conf = (float*)d_out + 2 * 65536;
  resize_k<<<(2 * 65536 + 255) / 256, 256, 0, stream>>>(flow, out_flow, 2, 32, 32, 256, 256, 8.0f);
  conv1s(C, hid, 128, 32, 32, 64, 3, 1, 1, conf_c1w, conf_c1b, cb1, ACT_RELU);
  conv1s(C, cb1, 64, 32, 32, 32, 3, 1, 1, conf_c2w, conf_c2b, cb2, ACT_RELU);
  conv1s(C, cb2, 32, 32, 32, 1, 1, 1, 0, conf_c3w, conf_c3b, cb3, ACT_SIG);
  resize_k<<<(65536 + 255) / 256, 256, 0, stream>>>(cb3, out_conf, 1, 32, 32, 256, 256, 1.0f);
}
